// StructureExtractor_7438883357611
// MI455X (gfx1250) — matrix operand order repaired and checked by reading, not compiled, not run on hardware
//
#include <hip/hip_runtime.h>

typedef __attribute__((ext_vector_type(16))) _Float16 v16h;
typedef __attribute__((ext_vector_type(8)))  float    v8f;
typedef __attribute__((ext_vector_type(4)))  unsigned int u32x4;
typedef __attribute__((ext_vector_type(8)))  int      i32x8;
typedef __attribute__((ext_vector_type(4)))  int      i32x4;

#define D 128
#define WT_PITCH 136   // padded LDS pitch (f16 elements), keeps 16B alignment

// ---------------------------------------------------------------------------
// Edge scatter: agg[dst[e]] += h[src[e]], 32 threads per edge, float4 per lane,
// fire-and-forget HW f32 atomics.
// ---------------------------------------------------------------------------
__global__ __launch_bounds__(256) void gin_scatter(const float* __restrict__ h,
                                                   const int* __restrict__ ei,
                                                   float* __restrict__ agg, int E) {
    int e = blockIdx.x * 8 + (threadIdx.x >> 5);   // wave-uniform edge id
    int lane = threadIdx.x & 31;
    if (e >= E) return;
    int s = ei[e];
    int d = ei[E + e];
    const float4 v = ((const float4*)(h + (size_t)s * D))[lane];
    float* ap = agg + (size_t)d * D + lane * 4;
    unsafeAtomicAdd(ap + 0, v.x);
    unsafeAtomicAdd(ap + 1, v.y);
    unsafeAtomicAdd(ap + 2, v.z);
    unsafeAtomicAdd(ap + 3, v.w);
}

// ---------------------------------------------------------------------------
// Segment pooling: out[seg[n], col_off + :] += feat[n, :]
// ---------------------------------------------------------------------------
__global__ __launch_bounds__(256) void gin_pool(const float* __restrict__ feat,
                                                const int* __restrict__ seg,
                                                float* __restrict__ out,
                                                int N, int col_off, int ostride) {
    int n = blockIdx.x * 8 + (threadIdx.x >> 5);
    int lane = threadIdx.x & 31;
    if (n >= N) return;
    int sg = seg[n];
    const float4 v = ((const float4*)(feat + (size_t)n * D))[lane];
    float* op = out + (size_t)sg * ostride + col_off + lane * 4;
    unsafeAtomicAdd(op + 0, v.x);
    unsafeAtomicAdd(op + 1, v.y);
    unsafeAtomicAdd(op + 2, v.z);
    unsafeAtomicAdd(op + 3, v.w);
}

// ---------------------------------------------------------------------------
// WMMA GEMM: out = act( (A0 [+ A1]) @ W + bias ), all [*,128] row-major f32.
// 256 threads = 8 waves; wave w owns rows [blk*128 + 16w, +16).
// W is DMA'd into LDS by the Tensor Data Mover (one tensor_load_to_lds,
// TENSORcnt-tracked), then converted f32->f16 transposed into the B-fragment
// layout so each B fragment is two contiguous 16-byte ds_load_b128.
// ---------------------------------------------------------------------------
template <bool HASADD, bool RELU>
__global__ __launch_bounds__(256) void gin_gemm(const float* __restrict__ A0,
                                                const float* __restrict__ A1,
                                                const float* __restrict__ W,
                                                const float* __restrict__ bias,
                                                float* __restrict__ out, int N) {
    __shared__ float    Wstage[D * D];     // raw f32 tile landed by TDM (64 KB)
    __shared__ _Float16 WT[D * WT_PITCH];  // transposed f16 B operand
    __shared__ float    bs[D];

    const int tid = threadIdx.x;

    // ---- TDM: DMA W [128x128] f32 from global into LDS (single issuing lane) ----
    if (tid == 0) {
        unsigned lds_off = (unsigned)(unsigned long long)(uintptr_t)&Wstage[0];
        unsigned long long ga = (unsigned long long)(uintptr_t)W;
        u32x4 g0;
        g0[0] = 1u;                                       // count=1, user descriptor
        g0[1] = lds_off;                                  // lds_addr (bytes)
        g0[2] = (unsigned)(ga & 0xffffffffu);             // global_addr[31:0]
        g0[3] = (unsigned)((ga >> 32) & 0x01ffffffu)      // global_addr[56:32]
                | (2u << 30);                             // type=2 ("image")
        i32x8 g1;
        g1[0] = (int)(2u << 16);     // data_size=2 -> 4-byte elements
        g1[1] = (int)(128u << 16);   // tensor_dim0[15:0] in bits [63:48]
        g1[2] = (int)(128u << 16);   // tensor_dim1[15:0] in bits [95:80]
        g1[3] = (int)(128u << 16);   // tile_dim0 in bits [127:112]
        g1[4] = 128;                 // tile_dim1 in bits [143:128]
        g1[5] = 128;                 // tensor_dim0_stride[31:0]
        g1[6] = 0;
        g1[7] = 0;
        i32x4 gz = {0, 0, 0, 0};     // groups 2/3 unused for 2-D tensors
#if __clang_major__ >= 23
        i32x8 gz8 = {0, 0, 0, 0, 0, 0, 0, 0};
        __builtin_amdgcn_tensor_load_to_lds(g0, g1, gz, gz, gz8, 0);
#else
        __builtin_amdgcn_tensor_load_to_lds(g0, g1, gz, gz, 0);
#endif
    }
    __builtin_amdgcn_s_wait_tensorcnt(0);
    __syncthreads();

    // ---- convert staged f32 -> f16, transposed with padded pitch ----
    #pragma unroll 4
    for (int i = tid; i < D * D; i += 256) {
        int n = i & (D - 1);
        int k = i >> 7;
        WT[n * WT_PITCH + k] = (_Float16)Wstage[k * D + n];
    }
    if (tid < D) bs[tid] = bias[tid];
    __syncthreads();

    const int wave = tid >> 5;
    const int lane = tid & 31;
    const int half = lane >> 4;   // 0 / 1
    const int lm   = lane & 15;

    const long long row0 = (long long)blockIdx.x * 128 + wave * 16;
    const long long arow = row0 + lm;           // row this lane reads for A
    const bool avalid = (arow < (long long)N);

    v8f acc[8] = {};

    #pragma unroll
    for (int kk = 0; kk < 4; ++kk) {
        const int kbase = kk * 32;

        // ---- A fragment: rows = lane%16, K split per ISA layout ----
        float4 f0 = {}, f1 = {}, f2 = {}, f3 = {};
        if (avalid) {
            const float* p = A0 + (size_t)arow * D + kbase + 8 * half;
            f0 = ((const float4*)p)[0];          // K = kbase+8h   .. +3
            f1 = ((const float4*)p)[1];          // K = kbase+8h+4 .. +7
            f2 = ((const float4*)(p + 16))[0];   // K = kbase+16+8h .. +3
            f3 = ((const float4*)(p + 16))[1];   // K = kbase+16+8h+4 ..
            if (HASADD) {
                const float* q = A1 + (size_t)arow * D + kbase + 8 * half;
                float4 g0 = ((const float4*)q)[0];
                float4 g1 = ((const float4*)q)[1];
                float4 g2 = ((const float4*)(q + 16))[0];
                float4 g3 = ((const float4*)(q + 16))[1];
                f0.x += g0.x; f0.y += g0.y; f0.z += g0.z; f0.w += g0.w;
                f1.x += g1.x; f1.y += g1.y; f1.z += g1.z; f1.w += g1.w;
                f2.x += g2.x; f2.y += g2.y; f2.z += g2.z; f2.w += g2.w;
                f3.x += g3.x; f3.y += g3.y; f3.z += g3.z; f3.w += g3.w;
            }
        }
        v16h a;
        a[0]  = (_Float16)f0.x; a[1]  = (_Float16)f0.y;
        a[2]  = (_Float16)f0.z; a[3]  = (_Float16)f0.w;
        a[4]  = (_Float16)f1.x; a[5]  = (_Float16)f1.y;
        a[6]  = (_Float16)f1.z; a[7]  = (_Float16)f1.w;
        a[8]  = (_Float16)f2.x; a[9]  = (_Float16)f2.y;
        a[10] = (_Float16)f2.z; a[11] = (_Float16)f2.w;
        a[12] = (_Float16)f3.x; a[13] = (_Float16)f3.y;
        a[14] = (_Float16)f3.z; a[15] = (_Float16)f3.w;

        // ---- 8 output tiles across N=128 ----
        #pragma unroll
        for (int nt = 0; nt < 8; ++nt) {
            const int ncol = nt * 16 + lm;
            const _Float16* bp = &WT[ncol * WT_PITCH + kbase + 8 * half];
            v16h b;
            #pragma unroll
            for (int j = 0; j < 8; ++j) { b[j] = bp[j]; b[j + 8] = bp[j + 16]; }
            acc[nt] = __builtin_amdgcn_wmma_f32_16x16x32_f16(
                false, a, false, b, (short)0, acc[nt], false, false);
        }
    }

    // ---- epilogue: bias (+ReLU), store f32 ----
    #pragma unroll
    for (int nt = 0; nt < 8; ++nt) {
        const int col = nt * 16 + lm;
        const float bv = bs[col];
        #pragma unroll
        for (int r = 0; r < 8; ++r) {
            long long row = row0 + r + 8 * half;
            if (row < (long long)N) {
                float v = acc[nt][r] + bv;
                if (RELU) v = fmaxf(v, 0.0f);
                out[(size_t)row * D + col] = v;
            }
        }
    }
}

// ---------------------------------------------------------------------------
extern "C" void kernel_launch(void* const* d_in, const int* in_sizes, int n_in,
                              void* d_out, int out_size, void* d_ws, size_t ws_size,
                              hipStream_t stream) {
    const float* x   = (const float*)d_in[0];
    const int*   ei  = (const int*)d_in[1];
    const int*   seg = (const int*)d_in[2];
    const float* W1  = (const float*)d_in[3];
    const float* b1  = (const float*)d_in[4];
    const float* W2  = (const float*)d_in[5];
    const float* b2  = (const float*)d_in[6];
    float* out = (float*)d_out;

    const int N = in_sizes[0] / D;            // 50000
    const int E = in_sizes[1] / 2;            // 800000
    const int L = in_sizes[3] / (D * D);      // 3
    const int OST = (L + 1) * D;              // 512

    float* agg = (float*)d_ws;
    float* z   = agg + (size_t)N * D;
    float* h   = z   + (size_t)N * D;

    hipMemsetAsync(d_out, 0, (size_t)out_size * sizeof(float), stream);

    const int scatter_blk = (E + 7) / 8;
    const int pool_blk    = (N + 7) / 8;
    const int gemm_blk    = (N + 127) / 128;

    // layer-0 input features pooled into columns [0,128)
    gin_pool<<<pool_blk, 256, 0, stream>>>(x, seg, out, N, 0, OST);

    const float* hcur = x;
    for (int l = 0; l < L; ++l) {
        hipMemsetAsync(agg, 0, (size_t)N * D * sizeof(float), stream);
        gin_scatter<<<scatter_blk, 256, 0, stream>>>(hcur, ei, agg, E);
        // z = relu((h + agg) @ W1[l] + b1[l])
        gin_gemm<true, true><<<gemm_blk, 256, 0, stream>>>(
            hcur, agg, W1 + (size_t)l * D * D, b1 + (size_t)l * D, z, N);
        // h = z @ W2[l] + b2[l]
        gin_gemm<false, false><<<gemm_blk, 256, 0, stream>>>(
            z, nullptr, W2 + (size_t)l * D * D, b2 + (size_t)l * D, h, N);
        // pool into columns [(l+1)*128, ...)
        gin_pool<<<pool_blk, 256, 0, stream>>>(h, seg, out, N, (l + 1) * D, OST);
        hcur = h;
    }
}
